// LRULayer_70815420776743
// MI455X (gfx1250) — matrix operand order repaired and checked by reading, not compiled, not run on hardware
//
#include <hip/hip_runtime.h>

// LRU layer for MI455X (gfx1250): bf16 WMMA GEMMs + chunked complex scan.
// L=8192, H=2048, N=1024.

#define LL 8192
#define HH 2048
#define NN 1024
#define T_CHUNK 128              // chunk length for the scan (= 2^7)
#define C_CHUNK (LL / T_CHUNK)   // 64 chunks

typedef __attribute__((ext_vector_type(8)))  __bf16 v8bf;
typedef __attribute__((ext_vector_type(16))) __bf16 v16bf;
typedef __attribute__((ext_vector_type(8)))  float  v8f;

// ---------------------------------------------------------------- helpers

__device__ __forceinline__ unsigned short f2bf(float f) {
  union { float f; unsigned u; } v; v.f = f;
  unsigned r = v.u + 0x7FFFu + ((v.u >> 16) & 1u);   // round-to-nearest-even
  return (unsigned short)(r >> 16);
}

// A-fragment for v_wmma_f32_16x16x32_bf16 (16x32 MxK, row-major source, ld = K-stride).
// Lane layout (ISA 7.12.2): lane m=lane&15, half=lane>>4;
// elems 0..7 -> K = k0 + 8*half + 0..7 ; elems 8..15 -> K = k0 + 16 + 8*half + 0..7
__device__ __forceinline__ v16bf frag_a(const unsigned short* __restrict__ base,
                                        int ld, int row, int k0, int half) {
  const unsigned short* p = base + (size_t)row * ld + (size_t)(k0 + 8 * half);
  v8bf lo = *reinterpret_cast<const v8bf*>(p);
  v8bf hi = *reinterpret_cast<const v8bf*>(p + 16);
  return __builtin_shufflevector(lo, hi, 0,1,2,3,4,5,6,7,8,9,10,11,12,13,14,15);
}

// B-fragment (32x16 KxN). Source stores output-column-major rows: src[col][k], ld = K-stride.
// Lane layout: col=lane&15, half=lane>>4; elems 0..7 -> K = k0 + 8*half + 0..7 ; elems 8..15 -> K = k0 + 16 + 8*half + 0..7
__device__ __forceinline__ v16bf frag_b(const unsigned short* __restrict__ base,
                                        int ld, int col, int k0, int half) {
  const unsigned short* p = base + (size_t)col * ld + (size_t)(k0 + 8 * half);
  v8bf lo = *reinterpret_cast<const v8bf*>(p);
  v8bf hi = *reinterpret_cast<const v8bf*>(p + 16);
  return __builtin_shufflevector(lo, hi, 0,1,2,3,4,5,6,7,8,9,10,11,12,13,14,15);
}

// ---------------------------------------------------------------- prep kernels

// Lambda = exp(-exp(nu) + i*exp(theta)); also Lambda^T_CHUNK via 7 complex squarings.
__global__ void lru_prep_params(const float* __restrict__ nu_log,
                                const float* __restrict__ theta_log,
                                float* __restrict__ lre, float* __restrict__ lim,
                                float* __restrict__ ltre, float* __restrict__ ltim) {
  int n = blockIdx.x * blockDim.x + threadIdx.x;
  if (n >= NN) return;
  float r  = expf(-expf(nu_log[n]));
  float th = expf(theta_log[n]);
  float ar = r * cosf(th);
  float ai = r * sinf(th);
  lre[n] = ar; lim[n] = ai;
  float tr = ar, ti = ai;
#pragma unroll
  for (int i = 0; i < 7; ++i) {         // Lambda^(2^7) = Lambda^128
    float nr = tr * tr - ti * ti;
    ti = 2.0f * tr * ti;
    tr = nr;
  }
  ltre[n] = tr; ltim[n] = ti;
}

// B_norm = (B_re + i B_im) * exp(gamma)[:,None]  ->  bf16, row-major (N,H)
__global__ void lru_prep_b(const float* __restrict__ B_re, const float* __restrict__ B_im,
                           const float* __restrict__ gamma_log,
                           unsigned short* __restrict__ bre, unsigned short* __restrict__ bim) {
  size_t i = (size_t)blockIdx.x * blockDim.x + threadIdx.x;
  if (i >= (size_t)NN * HH) return;
  int n = (int)(i / HH);
  float g = expf(gamma_log[n]);
  bre[i] = f2bf(B_re[i] * g);
  bim[i] = f2bf(B_im[i] * g);
}

// C -> bf16 (H,N); pre-negate C_im so GEMM2 is a single accumulate chain.
__global__ void lru_prep_c(const float* __restrict__ C_re, const float* __restrict__ C_im,
                           unsigned short* __restrict__ cre, unsigned short* __restrict__ cimn) {
  size_t i = (size_t)blockIdx.x * blockDim.x + threadIdx.x;
  if (i >= (size_t)HH * NN) return;
  cre[i]  = f2bf(C_re[i]);
  cimn[i] = f2bf(-C_im[i]);
}

// x -> bf16 (L,H)
__global__ void lru_prep_x(const float* __restrict__ x, unsigned short* __restrict__ xbf) {
  size_t i = (size_t)blockIdx.x * blockDim.x + threadIdx.x;
  if (i >= (size_t)LL * HH) return;
  xbf[i] = f2bf(x[i]);
}

// ---------------------------------------------------------------- GEMM1: Bu = x @ B_norm^T
// Block: 256 thr = 8 waves as 2(M) x 4(N); block tile 128x128; wave tile 64x32 per output.
// Dual (re,im) accumulators share the A fragments.
__global__ __launch_bounds__(256) void lru_gemm1(
    const unsigned short* __restrict__ xbf,   // (L,H) bf16
    const unsigned short* __restrict__ bre,   // (N,H) bf16
    const unsigned short* __restrict__ bim,   // (N,H) bf16
    float* __restrict__ bu_re, float* __restrict__ bu_im) {  // (L,N) f32
  const int lane = threadIdx.x & 31;
  const int wave = threadIdx.x >> 5;
  const int half = lane >> 4, lcol = lane & 15;
  const int rowBase = blockIdx.y * 128 + (wave >> 2) * 64;
  const int colBase = blockIdx.x * 128 + (wave & 3) * 32;

  v8f accr[4][2], acci[4][2];
#pragma unroll
  for (int m = 0; m < 4; ++m)
#pragma unroll
    for (int n = 0; n < 2; ++n) { accr[m][n] = (v8f)(0.0f); acci[m][n] = (v8f)(0.0f); }

  for (int k0 = 0; k0 < HH; k0 += 32) {
    if (k0 + 32 < HH)
      __builtin_prefetch(xbf + (size_t)(rowBase + lcol) * HH + k0 + 32, 0, 1);
    v16bf a[4];
#pragma unroll
    for (int m = 0; m < 4; ++m)
      a[m] = frag_a(xbf, HH, rowBase + m * 16 + lcol, k0, half);
    v16bf br[2], bi[2];
#pragma unroll
    for (int n = 0; n < 2; ++n) {
      br[n] = frag_b(bre, HH, colBase + n * 16 + lcol, k0, half);
      bi[n] = frag_b(bim, HH, colBase + n * 16 + lcol, k0, half);
    }
#pragma unroll
    for (int m = 0; m < 4; ++m)
#pragma unroll
      for (int n = 0; n < 2; ++n) {
        accr[m][n] = __builtin_amdgcn_wmma_f32_16x16x32_bf16(
            false, a[m], false, br[n], (short)0, accr[m][n], false, false);
        acci[m][n] = __builtin_amdgcn_wmma_f32_16x16x32_bf16(
            false, a[m], false, bi[n], (short)0, acci[m][n], false, false);
      }
  }
#pragma unroll
  for (int m = 0; m < 4; ++m)
#pragma unroll
    for (int n = 0; n < 2; ++n)
#pragma unroll
      for (int v = 0; v < 8; ++v) {
        int row = rowBase + m * 16 + v + 8 * half;   // C/D layout: VGPR v -> M = v + 8*half
        int col = colBase + n * 16 + lcol;
        bu_re[(size_t)row * NN + col] = accr[m][n][v];
        bu_im[(size_t)row * NN + col] = acci[m][n][v];
      }
}

// ---------------------------------------------------------------- scan (3 passes)

// Pass 1: per (chunk, channel), run the complex recurrence over T_CHUNK steps; keep only final.
__global__ void lru_scan_finals(const float* __restrict__ bu_re, const float* __restrict__ bu_im,
                                const float* __restrict__ lre, const float* __restrict__ lim,
                                float* __restrict__ fre, float* __restrict__ fim) {
  int g = blockIdx.x * blockDim.x + threadIdx.x;   // [0, C_CHUNK*NN)
  int c = g >> 10, n = g & (NN - 1);
  float ar = lre[n], ai = lim[n];
  float sr = 0.0f, si = 0.0f;
  size_t base = (size_t)(c * T_CHUNK) * NN + n;
  for (int t = 0; t < T_CHUNK; ++t) {
    float br = bu_re[base], bi = bu_im[base];
    float nsr = ar * sr - ai * si + br;
    si = ar * si + ai * sr + bi;
    sr = nsr;
    base += NN;
  }
  fre[(size_t)c * NN + n] = sr;
  fim[(size_t)c * NN + n] = si;
}

// Pass 2: serial prefix over 64 chunks per channel using Lambda^T_CHUNK.
__global__ void lru_scan_prefix(const float* __restrict__ fre, const float* __restrict__ fim,
                                const float* __restrict__ ltre, const float* __restrict__ ltim,
                                float* __restrict__ pre, float* __restrict__ pim) {
  int n = blockIdx.x * blockDim.x + threadIdx.x;
  if (n >= NN) return;
  float ar = ltre[n], ai = ltim[n];
  float pr = 0.0f, pi = 0.0f;
  for (int c = 0; c < C_CHUNK; ++c) {
    pre[(size_t)c * NN + n] = pr;
    pim[(size_t)c * NN + n] = pi;
    float fr = fre[(size_t)c * NN + n], fi = fim[(size_t)c * NN + n];
    float npr = ar * pr - ai * pi + fr;
    pi = ar * pi + ai * pr + fi;
    pr = npr;
  }
}

// Pass 3: re-run recurrence seeded with carry; emit states directly in bf16 (GEMM2 A-format).
__global__ void lru_scan_emit(const float* __restrict__ bu_re, const float* __restrict__ bu_im,
                              const float* __restrict__ lre, const float* __restrict__ lim,
                              const float* __restrict__ pre, const float* __restrict__ pim,
                              unsigned short* __restrict__ sre, unsigned short* __restrict__ sim) {
  int g = blockIdx.x * blockDim.x + threadIdx.x;
  int c = g >> 10, n = g & (NN - 1);
  float ar = lre[n], ai = lim[n];
  float sr = pre[(size_t)c * NN + n];
  float si = pim[(size_t)c * NN + n];
  size_t base = (size_t)(c * T_CHUNK) * NN + n;
  for (int t = 0; t < T_CHUNK; ++t) {
    float br = bu_re[base], bi = bu_im[base];
    float nsr = ar * sr - ai * si + br;
    si = ar * si + ai * sr + bi;
    sr = nsr;
    sre[base] = f2bf(sr);
    sim[base] = f2bf(si);
    base += NN;
  }
}

// ---------------------------------------------------------------- GEMM2: y = sre@Cre^T + sim@(-Cim)^T + x*D
__global__ __launch_bounds__(256) void lru_gemm2(
    const unsigned short* __restrict__ sre,   // (L,N) bf16
    const unsigned short* __restrict__ sim,   // (L,N) bf16
    const unsigned short* __restrict__ cre,   // (H,N) bf16
    const unsigned short* __restrict__ cimn,  // (H,N) bf16, pre-negated C_im
    const float* __restrict__ x, const float* __restrict__ D,
    float* __restrict__ y) {
  const int lane = threadIdx.x & 31;
  const int wave = threadIdx.x >> 5;
  const int half = lane >> 4, lcol = lane & 15;
  const int rowBase = blockIdx.y * 128 + (wave >> 2) * 64;   // over L
  const int colBase = blockIdx.x * 128 + (wave & 3) * 32;    // over H

  v8f acc[4][2];
#pragma unroll
  for (int m = 0; m < 4; ++m)
#pragma unroll
    for (int n = 0; n < 2; ++n) acc[m][n] = (v8f)(0.0f);

  for (int k0 = 0; k0 < NN; k0 += 32) {
    if (k0 + 32 < NN)
      __builtin_prefetch(sre + (size_t)(rowBase + lcol) * NN + k0 + 32, 0, 1);
    v16bf are[4], aim[4];
#pragma unroll
    for (int m = 0; m < 4; ++m) {
      are[m] = frag_a(sre, NN, rowBase + m * 16 + lcol, k0, half);
      aim[m] = frag_a(sim, NN, rowBase + m * 16 + lcol, k0, half);
    }
    v16bf br[2], bi[2];
#pragma unroll
    for (int n = 0; n < 2; ++n) {
      br[n] = frag_b(cre,  NN, colBase + n * 16 + lcol, k0, half);
      bi[n] = frag_b(cimn, NN, colBase + n * 16 + lcol, k0, half);
    }
#pragma unroll
    for (int m = 0; m < 4; ++m)
#pragma unroll
      for (int n = 0; n < 2; ++n) {
        acc[m][n] = __builtin_amdgcn_wmma_f32_16x16x32_bf16(
            false, are[m], false, br[n], (short)0, acc[m][n], false, false);
        acc[m][n] = __builtin_amdgcn_wmma_f32_16x16x32_bf16(
            false, aim[m], false, bi[n], (short)0, acc[m][n], false, false);
      }
  }
#pragma unroll
  for (int m = 0; m < 4; ++m)
#pragma unroll
    for (int n = 0; n < 2; ++n)
#pragma unroll
      for (int v = 0; v < 8; ++v) {
        int row = rowBase + m * 16 + v + 8 * half;
        int col = colBase + n * 16 + lcol;
        size_t idx = (size_t)row * HH + col;
        y[idx] = acc[m][n][v] + x[idx] * D[col];
      }
}

// ---------------------------------------------------------------- host

extern "C" void kernel_launch(void* const* d_in, const int* in_sizes, int n_in,
                              void* d_out, int out_size, void* d_ws, size_t ws_size,
                              hipStream_t stream) {
  (void)in_sizes; (void)n_in; (void)out_size; (void)ws_size;
  const float* x         = (const float*)d_in[0];
  const float* nu_log    = (const float*)d_in[1];
  const float* theta_log = (const float*)d_in[2];
  const float* B_re      = (const float*)d_in[3];
  const float* B_im      = (const float*)d_in[4];
  const float* C_re      = (const float*)d_in[5];
  const float* C_im      = (const float*)d_in[6];
  const float* D         = (const float*)d_in[7];
  const float* gamma_log = (const float*)d_in[8];
  float* y = (float*)d_out;

  char* w = (char*)d_ws;
  size_t off = 0;
  auto take = [&](size_t bytes) -> char* {
    char* p = w + off;
    off = (off + bytes + 255) & ~(size_t)255;
    return p;
  };
  unsigned short* xbf   = (unsigned short*)take((size_t)LL * HH * 2);   // 32 MB
  unsigned short* bre   = (unsigned short*)take((size_t)NN * HH * 2);   // 4 MB
  unsigned short* bim   = (unsigned short*)take((size_t)NN * HH * 2);   // 4 MB
  unsigned short* cre   = (unsigned short*)take((size_t)HH * NN * 2);   // 4 MB
  unsigned short* cimn  = (unsigned short*)take((size_t)HH * NN * 2);   // 4 MB
  float*          bu_re = (float*)take((size_t)LL * NN * 4);            // 32 MB
  float*          bu_im = (float*)take((size_t)LL * NN * 4);            // 32 MB
  unsigned short* sre   = (unsigned short*)take((size_t)LL * NN * 2);   // 16 MB
  unsigned short* sim   = (unsigned short*)take((size_t)LL * NN * 2);   // 16 MB
  float*          lre   = (float*)take(NN * 4);
  float*          lim   = (float*)take(NN * 4);
  float*          ltre  = (float*)take(NN * 4);
  float*          ltim  = (float*)take(NN * 4);
  float*          fre   = (float*)take((size_t)C_CHUNK * NN * 4);
  float*          fim   = (float*)take((size_t)C_CHUNK * NN * 4);
  float*          pref  = (float*)take((size_t)C_CHUNK * NN * 4);
  float*          pimf  = (float*)take((size_t)C_CHUNK * NN * 4);

  // prep
  lru_prep_params<<<(NN + 255) / 256, 256, 0, stream>>>(nu_log, theta_log, lre, lim, ltre, ltim);
  lru_prep_b<<<(int)(((size_t)NN * HH + 255) / 256), 256, 0, stream>>>(B_re, B_im, gamma_log, bre, bim);
  lru_prep_c<<<(int)(((size_t)HH * NN + 255) / 256), 256, 0, stream>>>(C_re, C_im, cre, cimn);
  lru_prep_x<<<(int)(((size_t)LL * HH + 255) / 256), 256, 0, stream>>>(x, xbf);

  // GEMM1: Bu = x @ B_norm^T   (grid: N/128 x L/128)
  lru_gemm1<<<dim3(NN / 128, LL / 128), 256, 0, stream>>>(xbf, bre, bim, bu_re, bu_im);

  // chunked complex scan
  lru_scan_finals<<<(C_CHUNK * NN) / 256, 256, 0, stream>>>(bu_re, bu_im, lre, lim, fre, fim);
  lru_scan_prefix<<<NN / 256, 256, 0, stream>>>(fre, fim, ltre, ltim, pref, pimf);
  lru_scan_emit<<<(C_CHUNK * NN) / 256, 256, 0, stream>>>(bu_re, bu_im, lre, lim, pref, pimf, sre, sim);

  // GEMM2: y = states@C^T (real part) + x*D   (grid: H/128 x L/128)
  lru_gemm2<<<dim3(HH / 128, LL / 128), 256, 0, stream>>>(sre, sim, cre, cimn, x, D, y);
}
